// CopulaModel_7181185319248
// MI455X (gfx1250) — compile-verified, matrix operand order checked by reading, not run on hardware
//
#include <hip/hip_runtime.h>
#include <hip/hip_bf16.h>
#include <math.h>

// ---------------------------------------------------------------------------
// CDNA5 (gfx1250) copula model.
//  Phase 1: stacked GRU, LDS-resident weights/state, WMMA bf16 matmuls with
//           fragment-order LDS layouts; f32 recurrence state in registers;
//           async global->LDS staging of the x tile.
//  Phase 2: linear heads (mu, softplus d, V).
//  Phase 3: Woodbury + det-lemma per-batch 64x64 SPD solve.
// ---------------------------------------------------------------------------

typedef __bf16 bf16_t;
typedef __attribute__((ext_vector_type(16))) __bf16 v16bf;
typedef __attribute__((ext_vector_type(8)))  float  v8f;

#define HID   64
#define G3    192          // 3*HID
#define TILE  64           // sequences per block
#define XSTR  68           // x tile stride (16B-aligned chunks, mult of 4)
#define NSEQ  8192         // B*T
#define TSTEPS 64

// Fragment-layout sizes (bf16 elements)
#define AFRAG_ELEMS (4 * 2 * 32 * 16)   // [mtile][ks][lane][16] = 4096
#define BFRAG_ELEMS (12 * 2 * 32 * 16)  // [ntile][ks][lane][16] = 12288

// f32 region: x tile + gX + gH2
static constexpr int GRU_F32_ELEMS = TILE * XSTR + 2 * TILE * G3;   // 28928
static constexpr int GRU_SMEM_BYTES =
    GRU_F32_ELEMS * 4 + 3 * BFRAG_ELEMS * 2 + 2 * AFRAG_ELEMS * 2;

static constexpr int SOLVE_F32_ELEMS =
    256 * 65 + 256 + 256 + 256 + 256 + 64 * 65 + 64;
static constexpr int SOLVE_SMEM_BYTES = SOLVE_F32_ELEMS * 4;

struct GruArgs {
    const float *xL, *xR;
    const float *L_Wih1, *L_Whh1, *L_bih1, *L_bhh1;
    const float *L_Wih2, *L_Whh2, *L_bih2, *L_bhh2;
    const float *R_Wih1, *R_Whh1, *R_bih1, *R_bhh1;
    const float *R_Wih2, *R_Whh2, *R_bih2, *R_bhh2;
    float *feat;   // [NSEQ][128]
};

__device__ __forceinline__ float sigmoidf_(float x) {
    return 1.0f / (1.0f + __expf(-x));
}

// tanh(x) = 1 - 2/(1 + e^{2x}) : one v_exp_f32, correct limits at +-inf.
__device__ __forceinline__ float tanhf_(float x) {
    return 1.0f - 2.0f / (1.0f + __expf(2.0f * x));
}

__device__ __forceinline__ v8f vzero8() {
    v8f z = {0.f, 0.f, 0.f, 0.f, 0.f, 0.f, 0.f, 0.f};
    return z;
}

// Pack two f32 into a dword of two bf16 (lo = a, hi = b).
__device__ __forceinline__ unsigned bfpack2(float a, float b) {
    __bf16 x = (__bf16)a, y = (__bf16)b;
    unsigned short xb, yb;
    __builtin_memcpy(&xb, &x, 2);
    __builtin_memcpy(&yb, &y, 2);
    return (unsigned)xb | ((unsigned)yb << 16);
}

// u32 index into an A-fragment buffer for the bf16 pair (r, j0), (r, j0+1),
// j0 even.  A layout (ISA 7.12.2, 16-bit A 16x32): lane<16 -> M=lane,
// K in {0..7,16..23}; lane>=16 -> M=lane-16, K in {8..15,24..31};
// element e -> k = (e<8 ? e : e+8) + (lanehi ? 8 : 0).
__device__ __forceinline__ int afrag_u32_idx(int r, int j0) {
    const int mtile = r >> 4, lr = r & 15;
    const int ks = j0 >> 5, jm = j0 & 31;
    const int lane = (((jm >> 3) & 1) << 4) + lr;
    const int e = (jm & 7) + ((jm & 16) ? 8 : 0);
    return ((mtile * 2 + ks) * 32 + lane) * 8 + (e >> 1);
}

// g[64][192] = h[64x64] x Wt[64x192], operands pre-swizzled in fragment order.
// 8 waves: wave w -> mtile = w&3, ntiles [(w>>2)*6, +6).
__device__ __forceinline__ void mm_frag(const v16bf *Af, const v16bf *Bf,
                                        float *gdst, int tid) {
    const int wave = tid >> 5;
    const int lane = tid & 31;
    const int mtile = wave & 3;
    const int nbase = (wave >> 2) * 6;

    v8f acc[6];
#pragma unroll
    for (int i = 0; i < 6; ++i) acc[i] = vzero8();

#pragma unroll
    for (int ks = 0; ks < 2; ++ks) {
        const v16bf a = Af[(mtile * 2 + ks) * 32 + lane];
#pragma unroll
        for (int nt = 0; nt < 6; ++nt) {
            const v16bf b = Bf[((nbase + nt) * 2 + ks) * 32 + lane];
            acc[nt] = __builtin_amdgcn_wmma_f32_16x16x32_bf16(
                false, a, false, b, (short)0, acc[nt], false, false);
        }
    }

    // C/D layout: VGPR g -> M = g (+8 for hi half-lanes), N = lane%16
    const int rbase = mtile * 16 + ((lane & 16) ? 8 : 0);
    const int cb = lane & 15;
#pragma unroll
    for (int nt = 0; nt < 6; ++nt) {
        const int col = (nbase + nt) * 16 + cb;
#pragma unroll
        for (int g = 0; g < 8; ++g)
            gdst[(rbase + g) * G3 + col] = acc[nt][g];
    }
}

__global__ __launch_bounds__(256) void gru_kernel(GruArgs ga) {
    extern __shared__ char smem[];
    float *xs    = (float *)smem;                  // [64][XSTR]
    float *gX    = xs + TILE * XSTR;               // [64][192]
    float *gH2   = gX + TILE * G3;                 // [64][192]
    bf16_t *Wf1  = (bf16_t *)(gH2 + TILE * G3);    // B-fragments Whh1^T
    bf16_t *WfI2 = Wf1 + BFRAG_ELEMS;              // B-fragments Wih2^T
    bf16_t *Wf2  = WfI2 + BFRAG_ELEMS;             // B-fragments Whh2^T
    bf16_t *hbf1 = Wf2 + BFRAG_ELEMS;              // A-fragments of h1
    bf16_t *hbf2 = hbf1 + AFRAG_ELEMS;             // A-fragments of h2
    unsigned *hbfu1 = (unsigned *)hbf1;
    unsigned *hbfu2 = (unsigned *)hbf2;

    const int tid  = threadIdx.x;
    const int side = blockIdx.y;                   // 0 = left, 1 = right
    const int n0   = blockIdx.x * TILE;

    const float *x     = side ? ga.xR : ga.xL;
    const float *Wih1p = side ? ga.R_Wih1 : ga.L_Wih1;
    const float *Whh1p = side ? ga.R_Whh1 : ga.L_Whh1;
    const float *bih1p = side ? ga.R_bih1 : ga.L_bih1;
    const float *bhh1p = side ? ga.R_bhh1 : ga.L_bhh1;
    const float *Wih2p = side ? ga.R_Wih2 : ga.L_Wih2;
    const float *Whh2p = side ? ga.R_Whh2 : ga.L_Whh2;
    const float *bih2p = side ? ga.R_bih2 : ga.L_bih2;
    const float *bhh2p = side ? ga.R_bhh2 : ga.L_bhh2;

    __builtin_prefetch(&x[(size_t)(n0 + (tid & 63)) * TSTEPS], 0, 3);

    // ---- x tile: async global->LDS copy (16B chunks, 16B-aligned dsts) ----
    {
#if defined(__gfx1250__)
        for (int i = tid; i < TILE * 16; i += 256) {
            const int r = i >> 4, c = (i & 15) * 4;
            const unsigned ldsoff =
                (unsigned)((const char *)(xs + r * XSTR + c) - (const char *)smem);
            const unsigned long long gaddr =
                (unsigned long long)(const void *)(x + (size_t)(n0 + r) * TSTEPS + c);
            asm volatile("global_load_async_to_lds_b128 %0, %1, off"
                         :: "v"(ldsoff), "v"(gaddr) : "memory");
        }
        asm volatile("s_wait_asynccnt 0x0" ::: "memory");
#else
        for (int i = tid; i < TILE * TSTEPS; i += 256) {
            const int r = i >> 6, c = i & 63;
            xs[r * XSTR + c] = x[(size_t)(n0 + r) * TSTEPS + c];
        }
#endif
    }

    // ---- stage weights f32 -> bf16 directly in B-fragment order ----
    // B layout: lane<16 -> N=lane, K in {0..7,16..23}; lane>=16 -> N=lane-16,
    // K in {8..15,24..31}; element e -> k = kb + (e<8 ? e : e+8).
    for (int i = tid; i < BFRAG_ELEMS; i += 256) {
        const int e = i & 15, frag = i >> 4;
        const int lane = frag & 31, t2 = frag >> 5;
        const int ks = t2 & 1, nt = t2 >> 1;
        const int col = nt * 16 + (lane & 15);     // gate output index 0..191
        const int k = ks * 32 + ((lane & 16) ? 8 : 0) + ((e < 8) ? e : e + 8);
        Wf1 [i] = (__bf16)Whh1p[col * HID + k];
        WfI2[i] = (__bf16)Wih2p[col * HID + k];
        Wf2 [i] = (__bf16)Whh2p[col * HID + k];
    }
    for (int i = tid; i < AFRAG_ELEMS / 2; i += 256) {
        hbfu1[i] = 0u; hbfu2[i] = 0u;
    }

    // gate threads own column pairs for the whole kernel:
    const int jp = tid & 31;
    const int j0 = jp * 2;
    const int rb = tid >> 5;          // 0..7

    // ---- hoist per-column weights/biases into registers (fixed per thread) --
    float w1r[2], w1z[2], w1n[2];       // layer-1 Wih (rank-1)
    float b1r[2], b1z[2], b1in[2], b1hn[2];
    float b2r[2], b2z[2], b2in[2], b2hn[2];
#pragma unroll
    for (int c = 0; c < 2; ++c) {
        const int j = j0 + c;
        w1r[c] = Wih1p[j];
        w1z[c] = Wih1p[64 + j];
        w1n[c] = Wih1p[128 + j];
        b1r[c] = bih1p[j] + bhh1p[j];
        b1z[c] = bih1p[64 + j] + bhh1p[64 + j];
        b1in[c] = bih1p[128 + j];
        b1hn[c] = bhh1p[128 + j];
        b2r[c] = bih2p[j] + bhh2p[j];
        b2z[c] = bih2p[64 + j] + bhh2p[64 + j];
        b2in[c] = bih2p[128 + j];
        b2hn[c] = bhh2p[128 + j];
    }

    // f32 recurrence state lives in registers (exclusive per-thread ownership)
    float h1r[8][2], h2r[8][2];
#pragma unroll
    for (int q = 0; q < 8; ++q) {
        h1r[q][0] = 0.f; h1r[q][1] = 0.f;
        h2r[q][0] = 0.f; h2r[q][1] = 0.f;
    }

    __syncthreads();

    const v16bf *Af1 = (const v16bf *)hbf1;
    const v16bf *Af2 = (const v16bf *)hbf2;
    const v16bf *Bf1 = (const v16bf *)Wf1;
    const v16bf *BfI2 = (const v16bf *)WfI2;
    const v16bf *Bf2 = (const v16bf *)Wf2;

    for (int t = 0; t < TSTEPS; ++t) {
        // ---- layer 1: gh1 = h1 @ Whh1^T ----
        mm_frag(Af1, Bf1, gX, tid);
        __syncthreads();

        // gate 1: gx1 is rank-1 broadcast x*Wih1 + bih1
#pragma unroll 2
        for (int q = 0; q < 8; ++q) {
            const int r = q * 8 + rb;
            const float xv = xs[r * XSTR + t];
            const float *g = gX + r * G3;
            float hn[2];
#pragma unroll
            for (int c = 0; c < 2; ++c) {
                const int j = j0 + c;
                const float a_r = xv * w1r[c] + b1r[c] + g[j];
                const float a_z = xv * w1z[c] + b1z[c] + g[64 + j];
                const float ghn = g[128 + j] + b1hn[c];
                const float a_n = xv * w1n[c] + b1in[c];
                const float rr = sigmoidf_(a_r);
                const float zz = sigmoidf_(a_z);
                const float nn = tanhf_(a_n + rr * ghn);
                hn[c] = (1.f - zz) * nn + zz * h1r[q][c];
                h1r[q][c] = hn[c];
            }
            hbfu1[afrag_u32_idx(r, j0)] = bfpack2(hn[0], hn[1]);
        }
        __syncthreads();

        // ---- layer 2: gx2 = h1 @ Wih2^T ; gh2 = h2 @ Whh2^T (separate,
        // n-gate needs tanh(gx_n + r * gh_n)) ----
        mm_frag(Af1, BfI2, gX, tid);
        mm_frag(Af2, Bf2, gH2, tid);
        __syncthreads();

#pragma unroll 2
        for (int q = 0; q < 8; ++q) {
            const int r = q * 8 + rb;
            const float *gav = gX + r * G3;
            const float *gbv = gH2 + r * G3;
            float hn[2];
#pragma unroll
            for (int c = 0; c < 2; ++c) {
                const int j = j0 + c;
                const float a_r = gav[j] + gbv[j] + b2r[c];
                const float a_z = gav[64 + j] + gbv[64 + j] + b2z[c];
                const float a_n = gav[128 + j] + b2in[c];
                const float g_n = gbv[128 + j] + b2hn[c];
                const float rr = sigmoidf_(a_r);
                const float zz = sigmoidf_(a_z);
                const float nn = tanhf_(a_n + rr * g_n);
                hn[c] = (1.f - zz) * nn + zz * h2r[q][c];
                h2r[q][c] = hn[c];
            }
            hbfu2[afrag_u32_idx(r, j0)] = bfpack2(hn[0], hn[1]);
        }
        __syncthreads();
    }

    // ---- emit features from registers: left -> cols [0,64), right [64,128) --
#pragma unroll
    for (int q = 0; q < 8; ++q) {
        const int r = q * 8 + rb;
        float *dst = ga.feat + (size_t)(n0 + r) * 128 + side * HID + j0;
        dst[0] = h2r[q][0];
        dst[1] = h2r[q][1];
    }
}

// ---------------------------------------------------------------------------
__global__ __launch_bounds__(256) void heads_kernel(
    const float *feat, const float *mean_w, const float *mean_b,
    const float *d_w, const float *d_b, const float *v_w, const float *v_b,
    float *mu, float *dvec, float *V) {
    const int gid = blockIdx.x * blockDim.x + threadIdx.x;
    if (gid >= NSEQ * 66) return;
    const int n = gid / 66;
    const int o = gid % 66;
    const float *f = feat + (size_t)n * 128;
    if (o < 64) {
        const float *w = v_w + o * 128;
        float s = 0.f;
#pragma unroll 8
        for (int k = 0; k < 128; ++k) s += f[k] * w[k];
        V[(size_t)n * 64 + o] = s + v_b[o];
    } else if (o == 64) {
        float s = 0.f;
#pragma unroll 8
        for (int k = 0; k < 128; ++k) s += f[k] * mean_w[k];
        mu[n] = s + mean_b[0];
    } else {
        float s = 0.f;
#pragma unroll 8
        for (int k = 0; k < 128; ++k) s += f[k] * d_w[k];
        s += d_b[0];
        dvec[n] = (s > 20.f) ? s : log1pf(__expf(s));  // softplus
    }
}

__global__ void zero_kernel(float *out) { out[0] = 0.f; }

// ---------------------------------------------------------------------------
// Woodbury: sigma = D + V V^T (V: 256x64).  M = I + V^T D^-1 V (64x64 SPD).
// quad  = r^T D^-1 r - || L^-1 (V^T D^-1 r) ||^2      (M = L L^T)
// logdet= 2*sum log L_ii + sum log d
// ---------------------------------------------------------------------------
__global__ __launch_bounds__(256) void solve_kernel(
    const float *V, const float *dvec, const float *y, const float *mu,
    float *out) {
    extern __shared__ char sm[];
    float *Vs   = (float *)sm;          // [256][65]
    float *invd = Vs + 256 * 65;        // [256]
    float *rv   = invd + 256;           // [256]
    float *red  = rv + 256;             // [256]  r^2/d partials
    float *red2 = red + 256;            // [256]  log d partials
    float *Ms   = red2 + 256;           // [64][65]
    float *wv   = Ms + 64 * 65;         // [64]

    const int b = blockIdx.x;
    const int tid = threadIdx.x;
    const int n0 = b * 256;

    for (int i = tid; i < 256 * 64; i += 256) {
        const int t = i >> 6, jj = i & 63;
        Vs[t * 65 + jj] = V[(size_t)(n0 + t) * 64 + jj];
    }
    {
        const float dd = dvec[n0 + tid];
        const float id = 1.f / dd;
        const float r = y[n0 + tid] - mu[n0 + tid];
        invd[tid] = id;
        rv[tid]   = r;
        red[tid]  = r * r * id;
        red2[tid] = __logf(dd);
    }
    __syncthreads();

    for (int e = tid; e < 64 * 64; e += 256) {
        const int i = e >> 6, jj = e & 63;
        float s = (i == jj) ? 1.f : 0.f;
        for (int t = 0; t < 256; ++t)
            s += Vs[t * 65 + i] * Vs[t * 65 + jj] * invd[t];
        Ms[i * 65 + jj] = s;
    }
    if (tid < 64) {
        float s = 0.f;
        for (int t = 0; t < 256; ++t)
            s += Vs[t * 65 + tid] * rv[t] * invd[t];
        wv[tid] = s;
    }
    __syncthreads();

    if (tid == 0) {
        float quad0 = 0.f, sld = 0.f;
        for (int t = 0; t < 256; ++t) { quad0 += red[t]; sld += red2[t]; }

        float logdetM = 0.f;
        for (int i = 0; i < 64; ++i) {
            float s = Ms[i * 65 + i];
            for (int k = 0; k < i; ++k) s -= Ms[i * 65 + k] * Ms[i * 65 + k];
            const float lii = sqrtf(fmaxf(s, 1e-20f));
            Ms[i * 65 + i] = lii;
            logdetM += 2.f * __logf(lii);
            const float inv = 1.f / lii;
            for (int r = i + 1; r < 64; ++r) {
                float s2 = Ms[r * 65 + i];
                for (int k = 0; k < i; ++k) s2 -= Ms[r * 65 + k] * Ms[i * 65 + k];
                Ms[r * 65 + i] = s2 * inv;
            }
        }
        float wMw = 0.f;
        for (int i = 0; i < 64; ++i) {
            float s = wv[i];
            for (int k = 0; k < i; ++k) s -= Ms[i * 65 + k] * wv[k];
            s /= Ms[i * 65 + i];
            wv[i] = s;
            wMw += s * s;
        }
        const float result = (quad0 - wMw) + (logdetM + sld);
        atomicAdd(out, result * (1.f / 32.f));
    }
}

// ---------------------------------------------------------------------------
extern "C" void kernel_launch(void *const *d_in, const int *in_sizes, int n_in,
                              void *d_out, int out_size, void *d_ws,
                              size_t ws_size, hipStream_t stream) {
    (void)in_sizes; (void)n_in; (void)out_size; (void)ws_size;

    const float *x_right = (const float *)d_in[0];
    const float *x_left  = (const float *)d_in[1];
    const float *y       = (const float *)d_in[2];
    const float *L_Wih1 = (const float *)d_in[3];
    const float *L_Whh1 = (const float *)d_in[4];
    const float *L_bih1 = (const float *)d_in[5];
    const float *L_bhh1 = (const float *)d_in[6];
    const float *L_Wih2 = (const float *)d_in[7];
    const float *L_Whh2 = (const float *)d_in[8];
    const float *L_bih2 = (const float *)d_in[9];
    const float *L_bhh2 = (const float *)d_in[10];
    const float *R_Wih1 = (const float *)d_in[11];
    const float *R_Whh1 = (const float *)d_in[12];
    const float *R_bih1 = (const float *)d_in[13];
    const float *R_bhh1 = (const float *)d_in[14];
    const float *R_Wih2 = (const float *)d_in[15];
    const float *R_Whh2 = (const float *)d_in[16];
    const float *R_bih2 = (const float *)d_in[17];
    const float *R_bhh2 = (const float *)d_in[18];
    const float *mean_w = (const float *)d_in[19];
    const float *mean_b = (const float *)d_in[20];
    const float *d_w    = (const float *)d_in[21];
    const float *d_b    = (const float *)d_in[22];
    const float *v_w    = (const float *)d_in[23];
    const float *v_b    = (const float *)d_in[24];

    float *feat = (float *)d_ws;                 // [8192][128]
    float *mu   = feat + (size_t)NSEQ * 128;     // [8192]
    float *dvec = mu + NSEQ;                     // [8192]
    float *V    = dvec + NSEQ;                   // [8192][64]

    (void)hipFuncSetAttribute((const void *)gru_kernel,
                              hipFuncAttributeMaxDynamicSharedMemorySize,
                              GRU_SMEM_BYTES);
    (void)hipFuncSetAttribute((const void *)solve_kernel,
                              hipFuncAttributeMaxDynamicSharedMemorySize,
                              SOLVE_SMEM_BYTES);

    GruArgs ga;
    ga.xL = x_left;  ga.xR = x_right;
    ga.L_Wih1 = L_Wih1; ga.L_Whh1 = L_Whh1; ga.L_bih1 = L_bih1; ga.L_bhh1 = L_bhh1;
    ga.L_Wih2 = L_Wih2; ga.L_Whh2 = L_Whh2; ga.L_bih2 = L_bih2; ga.L_bhh2 = L_bhh2;
    ga.R_Wih1 = R_Wih1; ga.R_Whh1 = R_Whh1; ga.R_bih1 = R_bih1; ga.R_bhh1 = R_bhh1;
    ga.R_Wih2 = R_Wih2; ga.R_Whh2 = R_Whh2; ga.R_bih2 = R_bih2; ga.R_bhh2 = R_bhh2;
    ga.feat = feat;

    gru_kernel<<<dim3(NSEQ / TILE, 2), 256, GRU_SMEM_BYTES, stream>>>(ga);

    const int headN = NSEQ * 66;
    heads_kernel<<<(headN + 255) / 256, 256, 0, stream>>>(
        feat, mean_w, mean_b, d_w, d_b, v_w, v_b, mu, dvec, V);

    zero_kernel<<<1, 1, 0, stream>>>((float *)d_out);

    solve_kernel<<<32, 256, SOLVE_SMEM_BYTES, stream>>>(
        V, dvec, y, mu, (float *)d_out);
}
